// MixedRationalQuadraticCouplingTransform_35167192219738
// MI455X (gfx1250) — matrix operand order repaired and checked by reading, not compiled, not run on hardware
//
#include <hip/hip_runtime.h>
#include <hip/hip_bf16.h>
#include <math.h>

typedef __attribute__((ext_vector_type(16))) __bf16 bf16x16;
typedef __attribute__((ext_vector_type(8)))  __bf16 bf16x8;
typedef __attribute__((ext_vector_type(8)))  float  f32x8;

#define N_SAMPLES 65536
#define FEAT      128
#define HIDDEN    512
#define CTX       64
#define NET_IN    160   // 2*32 (cos/sin) + 32 (identity regular) + 64 (context)
#define NET_OUT   992   // 32*15 + 32*16
#define NBINS     5
#define PI_F      3.14159265358979323846f
#define MIN_BW    0.001f
#define MIN_BH    0.001f
#define MIN_D     0.001f
#define TAILB     5.0f

// ---------------------------------------------------------------------------
// Weight repack: row-major W[K,N] (f32) -> bf16 fragments in WMMA B order.
// Fragment for tile (kt,nt), lane l, element e holds
//   W[kt*32 + 8*(l/16) + (e%8) + 16*(e/8), nt*16 + (l%16)]
// so the GEMM reads each per-lane fragment as one contiguous 32B chunk.
// ---------------------------------------------------------------------------
__global__ void pack_weight_kernel(const float* __restrict__ W,
                                   __bf16* __restrict__ Wp,
                                   int K, int Nout) {
    int idx = blockIdx.x * blockDim.x + threadIdx.x;
    int total = K * Nout;
    if (idx >= total) return;
    int e  = idx & 15;
    int l  = (idx >> 4) & 31;
    int t  = idx >> 9;           // kt*NT + nt
    int NT = Nout >> 4;
    int kt = t / NT;
    int nt = t - kt * NT;
    int row = kt * 32 + ((e >> 3) << 4) + ((l >> 4) << 3) + (e & 7);
    int col = nt * 16 + (l & 15);
    Wp[idx] = (__bf16)W[(size_t)row * Nout + col];
}

// ---------------------------------------------------------------------------
// Build net_in = [cos(ic) | sin(ic) | ir | context] in bf16, row-major [N,160]
// IDX_IC = cols 0..31, IDX_IR = cols 64..95 of the 128-wide input.
// ---------------------------------------------------------------------------
__global__ void build_netin_kernel(const float* __restrict__ in,
                                   const float* __restrict__ ctx,
                                   __bf16* __restrict__ A) {
    int idx = blockIdx.x * blockDim.x + threadIdx.x;
    if (idx >= N_SAMPLES * NET_IN) return;
    int c = idx % NET_IN;
    int s = idx / NET_IN;
    float v;
    if (c < 32)       v = cosf(in[(size_t)s * FEAT + c]);
    else if (c < 64)  v = sinf(in[(size_t)s * FEAT + (c - 32)]);
    else if (c < 96)  v = in[(size_t)s * FEAT + c];          // cols 64..95
    else              v = ctx[(size_t)s * CTX + (c - 96)];
    A[idx] = (__bf16)v;
}

// ---------------------------------------------------------------------------
// WMMA GEMM: C = act(A @ B + bias). A bf16 row-major [M,K]; B packed frags.
// Block = 256 threads = 8 waves arranged 2(M) x 4(N); wave tile = 64x64
// (4x4 WMMA 16x16 tiles); block tile = 128 x 256.
// A 16x32 bf16 fragment layout (ISA 7.12.2): lane l, elem e ->
//   K = k0 + 8*(l/16) + (e%8) + 16*(e/8)   (two contiguous 16B loads)
// C/D layout: lane l, vgpr r -> M = 8*(l/16)+r, N = l%16.
// Out-of-range N-tiles are clamped to tile NT-1 in the K loop (their WMMA
// results are garbage but discarded by the store-side column guard), keeping
// the inner loop free of predication/exec manipulation.
// ---------------------------------------------------------------------------
template <bool RELU, bool OUT_BF16>
__global__ __launch_bounds__(256)
void wmma_gemm_kernel(const __bf16* __restrict__ A,
                      const __bf16* __restrict__ Bp,
                      const float* __restrict__ bias,
                      void* __restrict__ Cout,
                      int M, int K, int Nout) {
    const int lane = threadIdx.x & 31;
    const int wave = threadIdx.x >> 5;
    const int wm   = wave & 1;
    const int wn   = wave >> 1;
    const int m0   = blockIdx.x * 128 + wm * 64;
    const int n0   = blockIdx.y * 256 + wn * 64;
    const int NT   = Nout >> 4;
    const int KT   = K >> 5;
    const int ln   = lane & 15;
    const int lh   = lane >> 4;      // 0 or 1
    const int ntile0 = n0 >> 4;

    // Clamped tile indices (K-invariant, hoisted out of the inner loop).
    int ntc[4];
    #pragma unroll
    for (int nt = 0; nt < 4; ++nt) {
        int t = ntile0 + nt;
        ntc[nt] = t < NT ? t : NT - 1;
    }

    f32x8 acc[4][4];
    #pragma unroll
    for (int nt = 0; nt < 4; ++nt) {
        int col = n0 + nt * 16 + ln;
        float bv = ((ntile0 + nt) < NT) ? bias[col] : 0.0f;
        #pragma unroll
        for (int mt = 0; mt < 4; ++mt)
            #pragma unroll
            for (int r = 0; r < 8; ++r)
                acc[mt][nt][r] = bv;
    }

    for (int kt = 0; kt < KT; ++kt) {
        const int kA = kt * 32 + lh * 8;

        bf16x16 afrag[4];
        #pragma unroll
        for (int mt = 0; mt < 4; ++mt) {
            const __bf16* ap = A + (size_t)(m0 + mt * 16 + ln) * K + kA;
            bf16x8 lo = *(const bf16x8*)ap;          // K = kA .. kA+7
            bf16x8 hi = *(const bf16x8*)(ap + 16);   // K = kA+16 .. kA+23
            afrag[mt] = __builtin_shufflevector(lo, hi,
                0, 1, 2, 3, 4, 5, 6, 7, 8, 9, 10, 11, 12, 13, 14, 15);
            // Speculative prefetch of next K-tile (unconditional: gfx1250
            // speculative prefetch drops bad translations silently, and the
            // overreach stays inside the workspace region).
            __builtin_prefetch(ap + 32, 0, 1);
        }

        bf16x16 bfrag[4];
        #pragma unroll
        for (int nt = 0; nt < 4; ++nt) {
            const __bf16* bp =
                Bp + (((size_t)(kt * NT + ntc[nt]) * 32) + lane) * 16;
            bfrag[nt] = *(const bf16x16*)bp;         // 32B contiguous
        }

        #pragma unroll
        for (int mt = 0; mt < 4; ++mt)
            #pragma unroll
            for (int nt = 0; nt < 4; ++nt)
                acc[mt][nt] = __builtin_amdgcn_wmma_f32_16x16x32_bf16(
                    false, afrag[mt], false, bfrag[nt],
                    (short)0, acc[mt][nt], false, false);
    }

    #pragma unroll
    for (int nt = 0; nt < 4; ++nt) {
        int col = n0 + nt * 16 + ln;
        if ((ntile0 + nt) < NT) {
            #pragma unroll
            for (int mt = 0; mt < 4; ++mt) {
                int rbase = m0 + mt * 16 + lh * 8;
                #pragma unroll
                for (int r = 0; r < 8; ++r) {
                    float v = acc[mt][nt][r];
                    if (RELU) v = v > 0.0f ? v : 0.0f;
                    if (OUT_BF16)
                        ((__bf16*)Cout)[(size_t)(rbase + r) * Nout + col] = (__bf16)v;
                    else
                        ((float*)Cout)[(size_t)(rbase + r) * Nout + col] = v;
                }
            }
        }
    }
}

// ---------------------------------------------------------------------------
// Rational-quadratic spline (nflows-style), forward.
// ---------------------------------------------------------------------------
__device__ __forceinline__ float softplusf(float x) {
    return (x > 20.0f) ? x : log1pf(expf(x));
}

__device__ void rq_spline(float x, const float* uw, const float* uh,
                          const float* ud, float left, float right,
                          float bottom, float top, float& y, float& lad) {
    const int K = NBINS;
    // widths: softmax + min-bin
    float mw = uw[0];
    for (int k = 1; k < K; ++k) mw = fmaxf(mw, uw[k]);
    float ew[NBINS], sw = 0.0f;
    for (int k = 0; k < K; ++k) { ew[k] = expf(uw[k] - mw); sw += ew[k]; }
    float cw[NBINS + 1];
    cw[0] = left;
    {
        float cum = 0.0f, span = right - left, inv = 1.0f / sw;
        for (int k = 0; k < K; ++k) {
            float w = MIN_BW + (1.0f - MIN_BW * K) * (ew[k] * inv);
            cum += w;
            cw[k + 1] = left + span * cum;
        }
        cw[K] = right;
    }
    // heights
    float mh = uh[0];
    for (int k = 1; k < K; ++k) mh = fmaxf(mh, uh[k]);
    float eh[NBINS], sh = 0.0f;
    for (int k = 0; k < K; ++k) { eh[k] = expf(uh[k] - mh); sh += eh[k]; }
    float ch[NBINS + 1];
    ch[0] = bottom;
    {
        float cum = 0.0f, span = top - bottom, inv = 1.0f / sh;
        for (int k = 0; k < K; ++k) {
            float h = MIN_BH + (1.0f - MIN_BH * K) * (eh[k] * inv);
            cum += h;
            ch[k + 1] = bottom + span * cum;
        }
        ch[K] = top;
    }
    // derivatives
    float dv[NBINS + 1];
    for (int k = 0; k <= K; ++k) dv[k] = MIN_D + softplusf(ud[k]);

    float xc = fminf(fmaxf(x, left), right);
    int idx = 0;
    for (int k = 1; k <= K; ++k) idx += (xc >= cw[k]) ? 1 : 0;
    if (idx > K - 1) idx = K - 1;

    float inw   = cw[idx + 1] - cw[idx];
    float inh   = ch[idx + 1] - ch[idx];
    float delta = inh / inw;
    float th    = (xc - cw[idx]) / inw;
    float omt   = 1.0f - th;
    float dk    = dv[idx], dk1 = dv[idx + 1];
    float num   = inh * (delta * th * th + dk * th * omt);
    float den   = delta + (dk + dk1 - 2.0f * delta) * th * omt;
    y = ch[idx] + num / den;
    float dnum = delta * delta *
                 (dk1 * th * th + 2.0f * delta * th * omt + dk * omt * omt);
    lad = logf(dnum) - 2.0f * logf(den);
}

// ---------------------------------------------------------------------------
// Spline epilogue: block = 4 samples x 64 transform dims; LDS logdet reduce.
// ---------------------------------------------------------------------------
__global__ __launch_bounds__(256)
void spline_kernel(const float* __restrict__ in,
                   const float* __restrict__ params,
                   float* __restrict__ out,
                   float* __restrict__ lad_out) {
    __shared__ float lred[256];
    const int tid = threadIdx.x;
    const int dim = tid & 63;
    const int sl  = tid >> 6;
    const int s   = blockIdx.x * 4 + sl;
    const float scale = 0.044194173824159216f;  // 1/sqrt(512)
    const float* prow = params + (size_t)s * NET_OUT;
    float lad = 0.0f;

    if (dim < 32) {
        // circular: params cc[d, 0:15], ud wraps; bounds +-pi
        int d = dim;
        float x = in[(size_t)s * FEAT + 32 + d];
        const float* base = prow + d * 15;
        float uw[NBINS], uh[NBINS], ud[NBINS + 1];
        for (int k = 0; k < NBINS; ++k) {
            uw[k] = base[k] * scale;
            uh[k] = base[NBINS + k] * scale;
            ud[k] = base[2 * NBINS + k];
        }
        ud[NBINS] = ud[0];
        float y;
        rq_spline(x, uw, uh, ud, -PI_F, PI_F, -PI_F, PI_F, y, lad);
        out[(size_t)s * FEAT + 32 + d] = y;
        out[(size_t)s * FEAT + d]      = in[(size_t)s * FEAT + d];       // ic
        out[(size_t)s * FEAT + 64 + d] = in[(size_t)s * FEAT + 64 + d];  // ir
    } else {
        // regular: params cr[d, 0:17]; bounds +-5 with identity tails
        int d = dim - 32;
        float x = in[(size_t)s * FEAT + 96 + d];
        const float* base = prow + 480 + d * 17;
        float uw[NBINS], uh[NBINS], ud[NBINS + 1];
        for (int k = 0; k < NBINS; ++k) {
            uw[k] = base[k] * scale;
            uh[k] = base[NBINS + k] * scale;
        }
        for (int k = 0; k <= NBINS; ++k) ud[k] = base[2 * NBINS + k];
        float y, l0;
        rq_spline(x, uw, uh, ud, -TAILB, TAILB, -TAILB, TAILB, y, l0);
        bool inside = (x >= -TAILB) && (x <= TAILB);
        if (!inside) { y = x; l0 = 0.0f; }
        lad = l0;
        out[(size_t)s * FEAT + 96 + d] = y;
    }

    lred[tid] = lad;
    __syncthreads();
    if (dim == 0) {
        float sum = 0.0f;
        for (int k = 0; k < 64; ++k) sum += lred[(sl << 6) + k];
        lad_out[s] = sum;
    }
}

// ---------------------------------------------------------------------------
extern "C" void kernel_launch(void* const* d_in, const int* in_sizes, int n_in,
                              void* d_out, int out_size, void* d_ws,
                              size_t ws_size, hipStream_t stream) {
    const float* inputs  = (const float*)d_in[0];
    const float* context = (const float*)d_in[1];
    const float* W1 = (const float*)d_in[2];
    const float* b1 = (const float*)d_in[3];
    const float* W2 = (const float*)d_in[4];
    const float* b2 = (const float*)d_in[5];
    const float* W3 = (const float*)d_in[6];
    const float* b3 = (const float*)d_in[7];

    char* ws = (char*)d_ws;
    size_t off = 0;
    auto take = [&](size_t bytes) -> char* {
        char* p = ws + off;
        off = (off + bytes + 255) & ~(size_t)255;
        return p;
    };
    __bf16* netin  = (__bf16*)take((size_t)N_SAMPLES * NET_IN * 2);
    __bf16* h1     = (__bf16*)take((size_t)N_SAMPLES * HIDDEN * 2);
    __bf16* h2     = (__bf16*)take((size_t)N_SAMPLES * HIDDEN * 2);
    float*  params = (float*) take((size_t)N_SAMPLES * NET_OUT * 4);
    __bf16* wp1    = (__bf16*)take((size_t)NET_IN * HIDDEN * 2);
    __bf16* wp2    = (__bf16*)take((size_t)HIDDEN * HIDDEN * 2);
    __bf16* wp3    = (__bf16*)take((size_t)HIDDEN * NET_OUT * 2);

    pack_weight_kernel<<<(NET_IN * HIDDEN + 255) / 256, 256, 0, stream>>>(
        W1, wp1, NET_IN, HIDDEN);
    pack_weight_kernel<<<(HIDDEN * HIDDEN + 255) / 256, 256, 0, stream>>>(
        W2, wp2, HIDDEN, HIDDEN);
    pack_weight_kernel<<<(HIDDEN * NET_OUT + 255) / 256, 256, 0, stream>>>(
        W3, wp3, HIDDEN, NET_OUT);

    build_netin_kernel<<<(N_SAMPLES * NET_IN + 255) / 256, 256, 0, stream>>>(
        inputs, context, netin);

    wmma_gemm_kernel<true, true>
        <<<dim3(N_SAMPLES / 128, HIDDEN / 256), 256, 0, stream>>>(
            netin, wp1, b1, h1, N_SAMPLES, NET_IN, HIDDEN);

    wmma_gemm_kernel<true, true>
        <<<dim3(N_SAMPLES / 128, HIDDEN / 256), 256, 0, stream>>>(
            h1, wp2, b2, h2, N_SAMPLES, HIDDEN, HIDDEN);

    wmma_gemm_kernel<false, false>
        <<<dim3(N_SAMPLES / 128, (NET_OUT + 255) / 256), 256, 0, stream>>>(
            h2, wp3, b3, params, N_SAMPLES, HIDDEN, NET_OUT);

    float* out = (float*)d_out;
    spline_kernel<<<N_SAMPLES / 4, 256, 0, stream>>>(
        inputs, params, out, out + (size_t)N_SAMPLES * FEAT);
}
